// VectorQuantizedVAE_78864189489765
// MI455X (gfx1250) — matrix operand order repaired and checked by reading, not compiled, not run on hardware
//
#include <hip/hip_runtime.h>

// ===================== CDNA5 WMMA types =====================
typedef __attribute__((ext_vector_type(16))) _Float16 v16h;
typedef __attribute__((ext_vector_type(8)))  float    v8f;
typedef __attribute__((ext_vector_type(4)))  float    v4f;

#define LEAKY(v) ((v) > 0.0f ? (v) : 0.01f * (v))

// =====================================================================
// Implicit-GEMM 3x3x3 conv (pad=1, stride 1 or 2), NCDHW, f16 WMMA.
// One wave computes a 16(M voxels) x 16(N cout) tile.
// K tiled as ONE INPUT CHANNEL PER 32-CHUNK (27 taps + 5 zero slots):
// slot->tap mapping is K-loop invariant, so offsets/validity are hoisted.
// All loads are UNCONDITIONAL from clamped-safe offsets; validity is
// applied as a post-load v_cndmask -> EXEC untouched, loads batch under
// one wait. Store epilogue: the 8 accumulators per lane are contiguous
// in memory, so when the M-tile is full (not at the 2^3 levels, where Mtot = 8)
// they go out as two global_store_b128.
// =====================================================================
__global__ __launch_bounds__(32)
void k_conv3_wmma(const float* __restrict__ x, const float* __restrict__ w,
                  float* __restrict__ y, const float* __restrict__ pbias,
                  int cin, int cout, int D, int H, int W,
                  int OD, int OH, int OW, int stride)
{
    const int l    = threadIdx.x;
    const int half = l >> 4;
    const int lm   = l & 15;
    const int m0   = blockIdx.x * 16;
    const int n0   = blockIdx.y * 16;
    const int Mtot = OD * OH * OW;
    const int Ktot = cin * 27;
    const int DHW  = D * H * W;
    const int n    = n0 + lm;
    const float in_bias = pbias ? pbias[0] : 0.0f;

    // geometry of this lane's A row
    const int m = m0 + lm;
    const bool mvalid = (m < Mtot);
    int oz = 0, oy = 0, ox = 0;
    if (mvalid) { oz = m / (OH * OW); int r = m % (OH * OW); oy = r / OW; ox = r % OW; }

    // ---- hoisted per-slot A/B geometry, with clamped-safe offsets ----
    int  aoff[16]; bool aval[16];
    int  boff[16]; bool bval[16];
#pragma unroll
    for (int v = 0; v < 8; ++v) {
#pragma unroll
        for (int e = 0; e < 2; ++e) {
            const int slot = 2 * v + e;
            {
                const int r = ((v < 4) ? (2 * v) : (16 + 2 * (v - 4))) + 8 * half + e;
                bool ok = mvalid && (r < 27);
                const int kz = r / 9, ryx = r % 9;
                const int ky = ryx / 3, kx = ryx % 3;
                const int iz = oz * stride + kz - 1;
                const int iy = oy * stride + ky - 1;
                const int ix = ox * stride + kx - 1;
                ok = ok && ((unsigned)iz < (unsigned)D) &&
                           ((unsigned)iy < (unsigned)H) &&
                           ((unsigned)ix < (unsigned)W);
                aval[slot] = ok;
                aoff[slot] = ok ? ((iz * H + iy) * W + ix) : 0;   // safe offset
            }
            {
                const int r = ((v < 4) ? (2 * v) : (16 + 2 * (v - 4))) + 8 * half + e;
                const bool ok = (n < cout) && (r < 27);
                bval[slot] = ok;
                boff[slot] = ok ? (n * Ktot + r) : 0;             // safe offset
            }
        }
    }

    v8f acc = {};
    for (int ci = 0; ci < cin; ++ci) {
        const float* xc = x + (size_t)ci * DHW;
        const float* wc = w + ci * 27;
        __builtin_prefetch(&wc[boff[0] + 27], 0, 1);  // speculative, safe
        v16h af, bf;
#pragma unroll
        for (int sl = 0; sl < 16; ++sl) {
            const float ar = xc[aoff[sl]];            // unconditional load
            const float br = wc[boff[sl]];            // unconditional load
            af[sl] = (_Float16)(aval[sl] ? (ar + in_bias) : 0.0f);
            bf[sl] = (_Float16)(bval[sl] ? br : 0.0f);
        }
        acc = __builtin_amdgcn_wmma_f32_16x16x32_f16(
            false, af, false, bf, (short)0, acc, false, false);
    }

    if (n < cout) {
        float* yp = y + (size_t)n * Mtot + m0 + 8 * half;   // lane-contiguous
        if (m0 + 16 <= Mtot) {
            // fast path (full tile; not taken when Mtot = 8): two 16B vector stores
            v4f s0 = { acc[0], acc[1], acc[2], acc[3] };
            v4f s1 = { acc[4], acc[5], acc[6], acc[7] };
            *(v4f*)(yp)     = s0;
            *(v4f*)(yp + 4) = s1;
        } else {
#pragma unroll
            for (int v = 0; v < 8; ++v) {
                const int mm = m0 + v + 8 * half;
                if (mm < Mtot) yp[v] = acc[v];
            }
        }
    }
}

// =====================================================================
// Transposed conv k=4, s=2, p=1 (torch ConvTranspose3d), weights (cin,cout,4,4,4).
// Output parity class (pz,py,px) = blockIdx.z fixes the 8-tap set, so
// K = cin*8 is tiled as FOUR CHANNELS PER 32-CHUNK (100% fill).
// Main loop over full 4-channel chunks is fully hoisted & branchless;
// a single tail chunk handles cin%4. Store epilogue takes one uniform
// branch on tile fullness to avoid per-element EXEC churn.
// =====================================================================
__global__ __launch_bounds__(32)
void k_convT4_wmma(const float* __restrict__ x, const float* __restrict__ w,
                   float* __restrict__ y, const float* __restrict__ pbias,
                   int cin, int cout, int D, int H, int W)
{
    const int l    = threadIdx.x;
    const int half = l >> 4;
    const int lm   = l & 15;
    const int m0   = blockIdx.x * 16;
    const int n0   = blockIdx.y * 16;
    const int cls  = blockIdx.z;
    const int pz = (cls >> 2) & 1, py = (cls >> 1) & 1, px = cls & 1;
    const int OD = 2 * D, OH = 2 * H, OW = 2 * W;
    const int Msub = D * H * W;
    const int DHW  = D * H * W;
    const int n    = n0 + lm;
    const float in_bias = pbias ? pbias[0] : 0.0f;

    const int m = m0 + lm;
    const bool mvalid = (m < Msub);
    int tz = 0, ty = 0, tx = 0;
    if (mvalid) { tz = m / (H * W); int r = m % (H * W); ty = r / W; tx = r % W; }
    const int oz = 2 * tz + pz, oy = 2 * ty + py, ox = 2 * tx + px;

    // ---- hoisted per-slot geometry: slot -> (local channel ciL, tap t) ----
    int  aoff[16]; bool aval[16]; int aciL[16];
    int  boff[16]; bool bval[16]; int bciL[16];
    const int cstride64 = cout * 64;
#pragma unroll
    for (int v = 0; v < 8; ++v) {
#pragma unroll
        for (int e = 0; e < 2; ++e) {
            const int slot = 2 * v + e;
            {
                const int r   = ((v < 4) ? (2 * v) : (16 + 2 * (v - 4))) + 8 * half + e;
                const int ciL = r >> 3, t = r & 7;
                const int kz = (1 - pz) + 2 * ((t >> 2) & 1);
                const int ky = (1 - py) + 2 * ((t >> 1) & 1);
                const int kx = (1 - px) + 2 * (t & 1);
                const int iz = (oz + 1 - kz) >> 1;
                const int iy = (oy + 1 - ky) >> 1;
                const int ix = (ox + 1 - kx) >> 1;
                const bool ok = mvalid &&
                                ((unsigned)iz < (unsigned)D) &&
                                ((unsigned)iy < (unsigned)H) &&
                                ((unsigned)ix < (unsigned)W);
                aval[slot] = ok;
                aciL[slot] = ciL;
                aoff[slot] = ok ? (ciL * DHW + (iz * H + iy) * W + ix) : 0;
            }
            {
                const int r   = ((v < 4) ? (2 * v) : (16 + 2 * (v - 4))) + 8 * half + e;
                const int ciL = r >> 3, t = r & 7;
                const int kz = (1 - pz) + 2 * ((t >> 2) & 1);
                const int ky = (1 - py) + 2 * ((t >> 1) & 1);
                const int kx = (1 - px) + 2 * (t & 1);
                const bool ok = (n < cout);
                bval[slot] = ok;
                bciL[slot] = ciL;
                boff[slot] = ok ? (ciL * cstride64 + n * 64 + kz * 16 + ky * 4 + kx) : 0;
            }
        }
    }

    v8f acc = {};
    const int cfull = cin & ~3;
    for (int ci0 = 0; ci0 < cfull; ci0 += 4) {
        const float* xc = x + (size_t)ci0 * DHW;
        const float* wc = w + (size_t)ci0 * cstride64;
        __builtin_prefetch(&wc[boff[0] + 4 * cstride64], 0, 1);
        v16h af, bf;
#pragma unroll
        for (int sl = 0; sl < 16; ++sl) {
            const float ar = xc[aoff[sl]];
            const float br = wc[boff[sl]];
            af[sl] = (_Float16)(aval[sl] ? (ar + in_bias) : 0.0f);
            bf[sl] = (_Float16)(bval[sl] ? br : 0.0f);
        }
        acc = __builtin_amdgcn_wmma_f32_16x16x32_f16(
            false, af, false, bf, (short)0, acc, false, false);
    }
    if (cfull < cin) {  // tail chunk: cin%4 channels, branchless via offset-select
        const int rem = cin - cfull;
        const float* xc = x + (size_t)cfull * DHW;
        const float* wc = w + (size_t)cfull * cstride64;
        v16h af, bf;
#pragma unroll
        for (int sl = 0; sl < 16; ++sl) {
            const bool av = aval[sl] && (aciL[sl] < rem);
            const bool bv = bval[sl] && (bciL[sl] < rem);
            const float ar = xc[av ? aoff[sl] : 0];
            const float br = wc[bv ? boff[sl] : 0];
            af[sl] = (_Float16)(av ? (ar + in_bias) : 0.0f);
            bf[sl] = (_Float16)(bv ? br : 0.0f);
        }
        acc = __builtin_amdgcn_wmma_f32_16x16x32_f16(
            false, af, false, bf, (short)0, acc, false, false);
    }

    if (n < cout) {
        const bool full = (m0 + 16 <= Msub);
        if (full) {
#pragma unroll
            for (int v = 0; v < 8; ++v) {
                const int mm = m0 + v + 8 * half;
                const int tz2 = mm / (H * W);
                const int r2  = mm % (H * W);
                const int ty2 = r2 / W, tx2 = r2 % W;
                y[((n * OD + (2 * tz2 + pz)) * OH + (2 * ty2 + py)) * OW +
                  (2 * tx2 + px)] = acc[v];
            }
        } else {
#pragma unroll
            for (int v = 0; v < 8; ++v) {
                const int mm = m0 + v + 8 * half;
                if (mm < Msub) {
                    const int tz2 = mm / (H * W);
                    const int r2  = mm % (H * W);
                    const int ty2 = r2 / W, tx2 = r2 % W;
                    y[((n * OD + (2 * tz2 + pz)) * OH + (2 * ty2 + py)) * OW +
                      (2 * tx2 + px)] = acc[v];
                }
            }
        }
    }
}

// =====================================================================
// nca: causal 2x2x2 average (pad low side with 0), optionally fusing a
// scalar pre-bias + leaky-ReLU on the pooled values (pad stays 0).
// =====================================================================
__global__ __launch_bounds__(256)
void k_nca(const float* __restrict__ x, float* __restrict__ y,
           const float* __restrict__ pbias, int do_leaky,
           int C, int D, int H, int W)
{
    const int i = blockIdx.x * blockDim.x + threadIdx.x;
    const int N = C * D * H * W;
    if (i >= N) return;
    const float bias = pbias ? pbias[0] : 0.0f;
    const int wv = i % W; int t = i / W;
    const int hv = t % H; t /= H;
    const int dv = t % D; const int c = t / D;
    const float* xc = x + (size_t)c * D * H * W;
    float s = 0.0f;
#pragma unroll
    for (int dz = 0; dz < 2; ++dz) {
        const int z = dv - 1 + dz; if (z < 0) continue;
#pragma unroll
        for (int dy = 0; dy < 2; ++dy) {
            const int yy = hv - 1 + dy; if (yy < 0) continue;
#pragma unroll
            for (int dx = 0; dx < 2; ++dx) {
                const int xx = wv - 1 + dx; if (xx < 0) continue;
                float v = xc[(z * H + yy) * W + xx] + bias;
                if (do_leaky) v = LEAKY(v);
                s += v;
            }
        }
    }
    y[i] = 0.125f * s;
}

// =====================================================================
// Fixup tail: y = leaky(a*scale + b2b + nca(b))  (skip-path nca fused)
// =====================================================================
__global__ __launch_bounds__(256)
void k_combine(const float* __restrict__ a, const float* __restrict__ b,
               float* __restrict__ y, const float* __restrict__ pscale,
               const float* __restrict__ pbias, int C, int D, int H, int W)
{
    const int i = blockIdx.x * blockDim.x + threadIdx.x;
    const int N = C * D * H * W;
    if (i >= N) return;
    const float scale = pscale[0];
    const float bias  = pbias[0];
    const int wv = i % W; int t = i / W;
    const int hv = t % H; t /= H;
    const int dv = t % D; const int c = t / D;
    const float* bc = b + (size_t)c * D * H * W;
    float s = 0.0f;
#pragma unroll
    for (int dz = 0; dz < 2; ++dz) {
        const int z = dv - 1 + dz; if (z < 0) continue;
#pragma unroll
        for (int dy = 0; dy < 2; ++dy) {
            const int yy = hv - 1 + dy; if (yy < 0) continue;
#pragma unroll
            for (int dx = 0; dx < 2; ++dx) {
                const int xx = wv - 1 + dx; if (xx < 0) continue;
                s += bc[(z * H + yy) * W + xx];
            }
        }
    }
    const float v = a[i] * scale + bias + 0.125f * s;
    y[i] = LEAKY(v);
}

// =====================================================================
// VQ: nearest neighbor over 256 codebook rows (first-min ties), emits q.
// =====================================================================
__global__ __launch_bounds__(256)
void k_vq(const float* __restrict__ x, const float* __restrict__ emb,
          float* __restrict__ q, int C, int Nvox, int nemb)
{
    const int i = blockIdx.x * blockDim.x + threadIdx.x;
    if (i >= Nvox) return;
    float v[32];
    for (int c = 0; c < C; ++c) v[c] = x[(size_t)c * Nvox + i];
    float best = 3.4e38f; int bi = 0;
    for (int e = 0; e < nemb; ++e) {
        const float* er = emb + (size_t)e * C;
        float d = 0.0f;
        for (int c = 0; c < C; ++c) { const float t = v[c] - er[c]; d += t * t; }
        if (d < best) { best = d; bi = e; }
    }
    const float* er = emb + (size_t)bi * C;
    for (int c = 0; c < C; ++c) q[(size_t)c * Nvox + i] = er[c];
}

// =====================================================================
// PixelShuffle3D(r=2) on (8,D,H,W) -> (1,2D,2H,2W), fusing per-channel bias.
// =====================================================================
__global__ __launch_bounds__(256)
void k_shuffle_bias(const float* __restrict__ x, const float* __restrict__ b,
                    float* __restrict__ y, int D, int H, int W)
{
    const int i = blockIdx.x * blockDim.x + threadIdx.x;
    const int OW = 2 * W, OH = 2 * H, OD = 2 * D;
    const int N = OD * OH * OW;
    if (i >= N) return;
    const int ow = i % OW; int t = i / OW;
    const int oh = t % OH; const int od = t / OH;
    const int rz = od & 1, d = od >> 1;
    const int ry = oh & 1, h = oh >> 1;
    const int rx = ow & 1, wv = ow >> 1;
    const int c = rz * 4 + ry * 2 + rx;
    y[i] = x[((size_t)c * D + d) * H * W + h * W + wv] + b[c];
}

// ===================== host-side orchestration =====================
namespace {

static inline const float* FP(void* const* d_in, int i) { return (const float*)d_in[i]; }

static void conv3(const float* x, const float* w, float* y, const float* pb,
                  int cin, int cout, int D, int H, int W, int stride,
                  hipStream_t s)
{
    const int OD = (stride == 2) ? D / 2 : D;
    const int OH = (stride == 2) ? H / 2 : H;
    const int OW = (stride == 2) ? W / 2 : W;
    const int Mtot = OD * OH * OW;
    dim3 g((Mtot + 15) / 16, (cout + 15) / 16, 1);
    k_conv3_wmma<<<g, 32, 0, s>>>(x, w, y, pb, cin, cout, D, H, W, OD, OH, OW, stride);
}

static void convT(const float* x, const float* w, float* y, const float* pb,
                  int cin, int cout, int D, int H, int W, hipStream_t s)
{
    const int Msub = D * H * W;
    dim3 g((Msub + 15) / 16, (cout + 15) / 16, 8);
    k_convT4_wmma<<<g, 32, 0, s>>>(x, w, y, pb, cin, cout, D, H, W);
}

static inline int ceil_div(int a, int b) { return (a + b - 1) / b; }

// Full fixup block. in -> out, temps B1/B2. mode: 0=level, 1=down, 2=up.
static void run_fixup(const float* in, float* out, float* B1, float* B2,
                      void* const* d_in, int spec, int cin, int cout, int mode,
                      int D, int H, int W, hipStream_t s)
{
    const int base = 1 + 8 * spec;
    const float* w1    = FP(d_in, base + 0);
    const float* wskip = FP(d_in, base + 1);
    const float* w2    = FP(d_in, base + 2);
    const float* b1a   = FP(d_in, base + 3);
    const float* b1b   = FP(d_in, base + 4);
    const float* b2a   = FP(d_in, base + 5);
    const float* b2b   = FP(d_in, base + 6);
    const float* scale = FP(d_in, base + 7);

    int OD, OH, OW;
    if (mode == 2)      { OD = 2 * D; OH = 2 * H; OW = 2 * W; }
    else if (mode == 1) { OD = D / 2; OH = H / 2; OW = W / 2; }
    else                { OD = D;     OH = H;     OW = W;     }

    // conv1(in + b1a) -> B1
    if (mode == 2) convT(in, w1, B1, b1a, cin, cout, D, H, W, s);
    else           conv3(in, w1, B1, b1a, cin, cout, D, H, W, (mode == 1) ? 2 : 1, s);
    // nca(leaky(B1 + b1b)) -> B2
    {
        const int N = cout * OD * OH * OW;
        k_nca<<<ceil_div(N, 256), 256, 0, s>>>(B1, B2, b1b, 1, cout, OD, OH, OW);
    }
    // conv2(B2 + b2a) -> B1
    conv3(B2, w2, B1, b2a, cout, cout, OD, OH, OW, 1, s);
    // skip conv(in + b1a) -> B2
    if (mode == 2) convT(in, wskip, B2, b1a, cin, cout, D, H, W, s);
    else           conv3(in, wskip, B2, b1a, cin, cout, D, H, W, (mode == 1) ? 2 : 1, s);
    // out = leaky(B1*scale + b2b + nca(B2))
    {
        const int N = cout * OD * OH * OW;
        k_combine<<<ceil_div(N, 256), 256, 0, s>>>(B1, B2, out, scale, b2b,
                                                   cout, OD, OH, OW);
    }
}

} // namespace

extern "C" void kernel_launch(void* const* d_in, const int* in_sizes, int n_in,
                              void* d_out, int out_size, void* d_ws, size_t ws_size,
                              hipStream_t stream)
{
    (void)in_sizes; (void)n_in; (void)out_size; (void)ws_size;

    const float* x = FP(d_in, 0);                 // (1,1,128,128,128)
    const float* q6_emb = FP(d_in, 153);          // (256,32)
    const float* q4_emb = FP(d_in, 154);          // (256,8)
    const float* q2_emb = FP(d_in, 155);          // (256,2)
    const float* d1_w   = FP(d_in, 156);          // (8,8,3,3,3)
    const float* d1_b   = FP(d_in, 157);          // (8,)
    float* out = (float*)d_out;

    // workspace layout (floats); MAXV = 4*128^3 = largest activation
    const size_t MAXV = (size_t)4 * 128 * 128 * 128;
    float* wsf = (float*)d_ws;
    float* B1  = wsf;
    float* B2  = wsf + MAXV;
    float* CH0 = wsf + 2 * MAXV;
    float* CH1 = wsf + 3 * MAXV;
    float* e2s = wsf + 4 * MAXV;                  // (16,32^3)  = 524288
    float* e4s = e2s + (size_t)16 * 32768;        // (64,8^3)   = 32768
    float* q6s = e4s + (size_t)64 * 512;          // (32,2^3)   = 256
    float* q4s = q6s + 256;                       // (8,8^3)    = 4096
    float* q2s = q4s + 4096;                      // (2,32^3)   = 65536

    const hipStream_t s = stream;
    const size_t F = sizeof(float);

    // ---------------- Encoder ----------------
    run_fixup(x,   CH0, B1, B2, d_in, 0, 1,   4,   0, 128, 128, 128, s); // e0
    run_fixup(CH0, CH1, B1, B2, d_in, 1, 4,   8,   1, 128, 128, 128, s); // e1 -> 64^3
    run_fixup(CH1, CH0, B1, B2, d_in, 2, 8,   16,  1, 64, 64, 64, s);    // e2 -> 32^3
    hipMemcpyAsync(e2s, CH0, (size_t)16 * 32768 * F, hipMemcpyDeviceToDevice, s);
    run_fixup(CH0, CH1, B1, B2, d_in, 3, 16,  32,  1, 32, 32, 32, s);    // e3 -> 16^3
    run_fixup(CH1, CH0, B1, B2, d_in, 4, 32,  64,  1, 16, 16, 16, s);    // e4 -> 8^3
    hipMemcpyAsync(e4s, CH0, (size_t)64 * 512 * F, hipMemcpyDeviceToDevice, s);
    run_fixup(CH0, CH1, B1, B2, d_in, 5, 64,  128, 1, 8, 8, 8, s);       // e5 -> 4^3
    run_fixup(CH1, CH0, B1, B2, d_in, 6, 128, 256, 1, 4, 4, 4, s);       // e6 -> 2^3

    // ---------------- Hierarchical quantization ----------------
    run_fixup(CH0, CH1, B1, B2, d_in, 7, 256, 32, 0, 2, 2, 2, s);        // q6b
    k_vq<<<1, 256, 0, s>>>(CH1, q6_emb, q6s, 32, 8, 256);                // q6
    run_fixup(q6s, CH0, B1, B2, d_in, 8, 32, 16, 2, 2, 2, 2, s);         // -> 4^3
    run_fixup(CH0, CH1, B1, B2, d_in, 9, 16, 8,  2, 4, 4, 4, s);         // u4 -> 8^3
    hipMemcpyAsync(CH0,            e4s, (size_t)64 * 512 * F, hipMemcpyDeviceToDevice, s);
    hipMemcpyAsync(CH0 + 64 * 512, CH1, (size_t)8 * 512 * F,  hipMemcpyDeviceToDevice, s);
    run_fixup(CH0, CH1, B1, B2, d_in, 10, 72, 8, 0, 8, 8, 8, s);         // q4b
    k_vq<<<ceil_div(512, 256), 256, 0, s>>>(CH1, q4_emb, q4s, 8, 512, 256); // q4
    run_fixup(q4s, CH0, B1, B2, d_in, 11, 8, 4, 2, 8, 8, 8, s);          // -> 16^3
    run_fixup(CH0, CH1, B1, B2, d_in, 12, 4, 2, 2, 16, 16, 16, s);       // u2 -> 32^3
    hipMemcpyAsync(CH0,               e2s, (size_t)16 * 32768 * F, hipMemcpyDeviceToDevice, s);
    hipMemcpyAsync(CH0 + 16 * 32768,  CH1, (size_t)2 * 32768 * F,  hipMemcpyDeviceToDevice, s);
    run_fixup(CH0, CH1, B1, B2, d_in, 13, 18, 2, 0, 32, 32, 32, s);      // q2b
    k_vq<<<ceil_div(32768, 256), 256, 0, s>>>(CH1, q2_emb, q2s, 2, 32768, 256); // q2

    // ---------------- Decoder ----------------
    run_fixup(q6s, CH0, B1, B2, d_in, 14, 32,  128, 2, 2, 2, 2, s);      // d6 -> 4^3
    run_fixup(CH0, CH1, B1, B2, d_in, 15, 128, 64,  2, 4, 4, 4, s);      // d5 -> 8^3
    hipMemcpyAsync(CH0,            CH1, (size_t)64 * 512 * F, hipMemcpyDeviceToDevice, s);
    hipMemcpyAsync(CH0 + 64 * 512, q4s, (size_t)8 * 512 * F,  hipMemcpyDeviceToDevice, s);
    run_fixup(CH0, CH1, B1, B2, d_in, 16, 72, 32, 2, 8, 8, 8, s);        // d4 -> 16^3
    run_fixup(CH1, CH0, B1, B2, d_in, 17, 32, 16, 2, 16, 16, 16, s);     // d3 -> 32^3
    hipMemcpyAsync(CH1,              CH0, (size_t)16 * 32768 * F, hipMemcpyDeviceToDevice, s);
    hipMemcpyAsync(CH1 + 16 * 32768, q2s, (size_t)2 * 32768 * F,  hipMemcpyDeviceToDevice, s);
    run_fixup(CH1, CH0, B1, B2, d_in, 18, 18, 8, 2, 32, 32, 32, s);      // d2 -> 64^3

    // SubPixelConvolution3D: conv3d(8->8) + bias + PixelShuffle(2) + nca
    conv3(CH0, d1_w, CH1, nullptr, 8, 8, 64, 64, 64, 1, s);
    {
        const int N = 8 * 64 * 64 * 64; // = 128^3 output voxels
        k_shuffle_bias<<<ceil_div(N, 256), 256, 0, s>>>(CH1, d1_b, CH0, 64, 64, 64);
        k_nca<<<ceil_div(N, 256), 256, 0, s>>>(CH0, out, nullptr, 0, 1, 128, 128, 128);
    }
}
